// CausalSelfAttention_41455024341746
// MI455X (gfx1250) — matrix operand order repaired and checked by reading, not compiled, not run on hardware
//
#include <hip/hip_runtime.h>

// ---------------------------------------------------------------------------
// CausalSelfAttention for MI455X (gfx1250, wave32, WMMA f16 16x16x32 + TDM)
//   B=64 T=256 C=384 H=6 D=64   qkv width = 1152
// Pipeline:
//   cvt(x->f16), transpose(Wqkv), transpose(Wproj)          (one-time, tiny)
//   qkvh = f16( xh @ WqkvT + b )      TDM-staged WMMA GEMM, double buffered
//   vtg  = per-head transpose of V                          (LDS-tiled)
//   atth = flash-attention(qkvh, vtg) TDM-staged, K/V double buffered
//   out  = atth @ WprojT + b  (f32)   TDM-staged WMMA GEMM
// ---------------------------------------------------------------------------

typedef _Float16 v4h  __attribute__((ext_vector_type(4)));
typedef _Float16 v8h  __attribute__((ext_vector_type(8)));
typedef _Float16 v16h __attribute__((ext_vector_type(16)));
typedef float    v8f  __attribute__((ext_vector_type(8)));
typedef unsigned int sa_u32x4 __attribute__((ext_vector_type(4)));
typedef int          sa_i32x8 __attribute__((ext_vector_type(8)));
typedef int          sa_i32x4 __attribute__((ext_vector_type(4)));

#define SA_B 64
#define SA_T 256
#define SA_C 384
#define SA_H 6
#define SA_D 64
#define SA_QKVN 1152
#define NEG_BIG (-3.0e38f)

// ---------------------------------------------------------------------------
// Tensor Data Mover: 2D f16 tile (tile_w elems per row, tile_h rows) from
// global (row stride = row_stride_elems) into LDS, padding each row by
// 8 halves (so LDS row pitch = tile_w + 8, keeping 16B alignment per row).
// D# layout per CDNA5 ISA (group0 128b, group1 256b).
// pad_int_code: 3 -> pad every 16 DWORDs (32-half rows), 4 -> every 32 DWORDs
// pad_amt_code: 3 -> insert 4 DWORDs (8 halves)
// ---------------------------------------------------------------------------
__device__ __forceinline__ void sa_tdm_load_2d(
    void* lds_ptr, const void* gptr,
    unsigned tile_w, unsigned tile_h,
    unsigned long long row_stride_elems,
    unsigned pad_int_code, unsigned pad_amt_code)
{
#if __has_builtin(__builtin_amdgcn_tensor_load_to_lds)
    const unsigned long long ga = (unsigned long long)gptr;
    sa_u32x4 g0;
    g0[0] = 1u;                                            // count=1, user D#
    g0[1] = (unsigned)(unsigned long long)lds_ptr;         // lds_addr (low 32 = LDS offset)
    g0[2] = (unsigned)(ga & 0xffffffffu);                  // global_addr[31:0]
    g0[3] = (unsigned)((ga >> 32) & 0x01ffffffu) | (2u << 30); // addr[56:32] | type=2
    sa_i32x8 g1;
    g1[0] = (int)((1u << 16)                               // data_size = 2 bytes
                | (1u << 20)                               // pad_enable
                | (pad_int_code << 22) | (pad_amt_code << 25));
    g1[1] = (int)((tile_w & 0xffffu) << 16);               // tensor_dim0[15:0]
    g1[2] = (int)(((tile_w >> 16) & 0xffffu) | ((tile_h & 0xffffu) << 16)); // dim0 hi | dim1 lo
    g1[3] = (int)(((tile_h >> 16) & 0xffffu) | ((tile_w & 0xffffu) << 16)); // dim1 hi | tile_dim0
    g1[4] = (int)(tile_h & 0xffffu);                       // tile_dim1 (tile_dim2 = 0)
    g1[5] = (int)(row_stride_elems & 0xffffffffull);       // tensor_dim0_stride lo
    g1[6] = (int)((row_stride_elems >> 32) & 0xffffull);   // stride hi (dim1_stride = 0)
    g1[7] = 0;
    const sa_i32x4 z4 = {0, 0, 0, 0};
    const sa_i32x8 z8 = {0, 0, 0, 0, 0, 0, 0, 0};
    __builtin_amdgcn_tensor_load_to_lds(g0, g1, z4, z4, z8, 0);
#else
    // fallback: calling wave (32 lanes) copies v8h chunks
    const int lane = threadIdx.x & 31;
    _Float16* lds = (_Float16*)lds_ptr;
    const _Float16* g = (const _Float16*)gptr;
    const unsigned pitch = tile_w + 8u;
    for (unsigned r = 0; r < tile_h; ++r)
        for (unsigned c = lane * 8u; c < tile_w; c += 256u)
            *(v8h*)(lds + (size_t)r * pitch + c) =
                *(const v8h*)(g + (size_t)r * row_stride_elems + c);
#endif
}

__device__ __forceinline__ void sa_wait_tdm() {
#if __has_builtin(__builtin_amdgcn_tensor_load_to_lds)
    __builtin_amdgcn_s_wait_tensorcnt(0);
#endif
}

// A-fragment (16x32 f16): lane L = row M=L%16,
// halves i: K = i + (L<16?0:16)  => two contiguous 16B LDS reads.
__device__ __forceinline__ v16h sa_load_afrag(const _Float16* rowp, int koff) {
    v8h lo = *(const v8h*)(rowp + 2 * koff);
    v8h hi = *(const v8h*)(rowp + 2 * koff + 8);
    return __builtin_shufflevector(lo, hi, 0,1,2,3,4,5,6,7,8,9,10,11,12,13,14,15);
}

// ---------------------------------------------------------------------------
// One-time prep kernels (operate on <13 MB total; negligible vs. main work)
// ---------------------------------------------------------------------------
__global__ __launch_bounds__(256) void sa_cvt(
    const float* __restrict__ in, _Float16* __restrict__ out, int n4)
{
    const int i = blockIdx.x * 256 + threadIdx.x;
    if (i < n4) {
        const float4 f = *(const float4*)(in + (size_t)i * 4);
        v4h hv = { (_Float16)f.x, (_Float16)f.y, (_Float16)f.z, (_Float16)f.w };
        *(v4h*)(out + (size_t)i * 4) = hv;
    }
}

// Wt[n][k] = (f16) W[k][n]   (W is [K][N] row-major)
__global__ __launch_bounds__(256) void sa_wtrans(
    const float* __restrict__ W, _Float16* __restrict__ Wt, int K, int N)
{
    const int i = blockIdx.x * 256 + threadIdx.x;
    if (i < N * K) {
        const int n = i / K, k = i - n * K;
        Wt[i] = (_Float16)W[(size_t)k * N + n];
    }
}

// vtg[(bh*64+d)*256 + t] = qkvh[(b*256+t)*1152 + 768 + h*64 + d]  (LDS-tiled)
__global__ __launch_bounds__(128) void sa_vtrans(
    const _Float16* __restrict__ qkvh, _Float16* __restrict__ vtg)
{
    __shared__ _Float16 Ts[64][72];
    const int t = threadIdx.x;
    const int bh = blockIdx.x, kt = blockIdx.y;
    const int b = bh / SA_H, h = bh % SA_H;
    const _Float16* vg = qkvh + (size_t)b * SA_T * SA_QKVN + 2 * SA_H * SA_D + h * SA_D;
    const int colg = (t & 15) * 4, rbase = t >> 4;
    #pragma unroll
    for (int p = 0; p < 8; ++p) {
        const int row = p * 8 + rbase;                   // key within tile
        *(v4h*)&Ts[row][colg] =
            *(const v4h*)(vg + (size_t)(kt * 64 + row) * SA_QKVN + colg);
    }
    __syncthreads();
    #pragma unroll
    for (int p = 0; p < 8; ++p) {
        const int d = p * 8 + rbase;
        v4h ov = { Ts[colg + 0][d], Ts[colg + 1][d], Ts[colg + 2][d], Ts[colg + 3][d] };
        *(v4h*)(vtg + ((size_t)bh * 64 + d) * SA_T + kt * 64 + colg) = ov;
    }
}

// ---------------------------------------------------------------------------
// GEMM: C[M,N] = A[M,K](f16) * Bt[N,K](f16, pre-transposed) + bias[N]
// 128x64 tile / 128-thread block; wave w: rows [w*32, w*32+32) -> 8 WMMA/step.
// TDM staging, double buffered across K-steps of 32.
// ---------------------------------------------------------------------------
template <typename OutT>
__global__ __launch_bounds__(128) void sa_gemm_wmma(
    const _Float16* __restrict__ A, const _Float16* __restrict__ Bt,
    const float* __restrict__ bias, OutT* __restrict__ Cmat,
    int M, int N, int K)
{
    __shared__ _Float16 As[2][128][40];   // [m][k] pitch 40 halves (80B)
    __shared__ _Float16 Bs[2][64][40];    // [n][k]
    const int t = threadIdx.x, lane = t & 31, w = t >> 5;
    const int hh = lane >> 4, ln = lane & 15;
    const int m0 = blockIdx.y * 128, n0 = blockIdx.x * 64;
    const int koff = hh * 8, koff2 = hh * 16;

    v8f acc[2][4];
    #pragma unroll
    for (int rt = 0; rt < 2; ++rt)
        #pragma unroll
        for (int nt = 0; nt < 4; ++nt)
            acc[rt][nt] = (v8f){0.f,0.f,0.f,0.f,0.f,0.f,0.f,0.f};

    if (w == 0) {
        sa_tdm_load_2d(&As[0][0][0], A + (size_t)m0 * K, 32, 128, (unsigned long long)K, 3, 3);
        sa_tdm_load_2d(&Bs[0][0][0], Bt + (size_t)n0 * K, 32, 64, (unsigned long long)K, 3, 3);
        sa_wait_tdm();
    }
    __syncthreads();

    const int nsteps = K / 32;
    for (int s = 0; s < nsteps; ++s) {
        const int cur = s & 1;
        if (s + 1 < nsteps && w == 0) {   // async prefetch of next K tile
            sa_tdm_load_2d(&As[1 - cur][0][0], A + (size_t)m0 * K + (s + 1) * 32,
                           32, 128, (unsigned long long)K, 3, 3);
            sa_tdm_load_2d(&Bs[1 - cur][0][0], Bt + (size_t)n0 * K + (s + 1) * 32,
                           32, 64, (unsigned long long)K, 3, 3);
        }
        // batch fragment loads, then WMMA chain
        const v16h a0 = sa_load_afrag(&As[cur][w * 32 + ln][0], koff);
        const v16h a1 = sa_load_afrag(&As[cur][w * 32 + 16 + ln][0], koff);
        v16h bf[4];
        #pragma unroll
        for (int nt = 0; nt < 4; ++nt)
            bf[nt] = *(const v16h*)&Bs[cur][nt * 16 + ln][koff2];
        #pragma unroll
        for (int nt = 0; nt < 4; ++nt)
            acc[0][nt] = __builtin_amdgcn_wmma_f32_16x16x32_f16(
                false, a0, false, bf[nt], (short)0, acc[0][nt], false, false);
        #pragma unroll
        for (int nt = 0; nt < 4; ++nt)
            acc[1][nt] = __builtin_amdgcn_wmma_f32_16x16x32_f16(
                false, a1, false, bf[nt], (short)0, acc[1][nt], false, false);
        if (s + 1 < nsteps) {
            if (w == 0) sa_wait_tdm();
            __syncthreads();
        }
    }

    #pragma unroll
    for (int rt = 0; rt < 2; ++rt)
        #pragma unroll
        for (int nt = 0; nt < 4; ++nt) {
            const int col = n0 + nt * 16 + ln;
            const float bn = bias[col];
            #pragma unroll
            for (int r = 0; r < 8; ++r) {
                const int row = m0 + w * 32 + rt * 16 + r + hh * 8;
                Cmat[(size_t)row * N + col] = (OutT)(acc[rt][nt][r] + bn);
            }
        }
}

// ---------------------------------------------------------------------------
// Flash attention: block = (b*H+h, 64-row q tile); wave w owns 16 q rows.
// Q/K/V tiles TDM-staged (raw f16 byte moves); K/V double buffered across
// the causal key loop; 1/sqrt(D) applied to S after WMMA.
// ---------------------------------------------------------------------------
__global__ __launch_bounds__(128) void sa_attn_wmma(
    const _Float16* __restrict__ qkvh,   // [B*T][1152]
    const _Float16* __restrict__ vtg,    // [B*H*64][256]  (V transposed per head)
    _Float16* __restrict__ att)          // [B*T][384]
{
    __shared__ _Float16 Qs[64][72];
    __shared__ _Float16 Ks[2][64][72];   // [key][d]
    __shared__ _Float16 Vt[2][64][72];   // [d][key]
    __shared__ _Float16 Ps[64][72];      // [q][key]
    const int t = threadIdx.x, lane = t & 31, w = t >> 5;
    const int hh = lane >> 4, ln = lane & 15;
    const int bh = blockIdx.x, b = bh / SA_H, h = bh % SA_H;
    const int qt = blockIdx.y, q0 = qt * 64;
    const int koff = hh * 8, koff2 = hh * 16;

    const _Float16* qg = qkvh + (size_t)b * SA_T * SA_QKVN + h * SA_D;
    const _Float16* kg = qg + SA_H * SA_D;
    const _Float16* vg = vtg + (size_t)bh * 64 * SA_T;

    if (w == 0) {
        sa_tdm_load_2d(&Qs[0][0],    qg + (size_t)q0 * SA_QKVN, 64, 64, SA_QKVN, 4, 3);
        sa_tdm_load_2d(&Ks[0][0][0], kg,                        64, 64, SA_QKVN, 4, 3);
        sa_tdm_load_2d(&Vt[0][0][0], vg,                        64, 64, SA_T,    4, 3);
        sa_wait_tdm();
    }
    __syncthreads();

    v16h qa[2];
    #pragma unroll
    for (int c = 0; c < 2; ++c)
        qa[c] = sa_load_afrag(&Qs[w * 16 + ln][c * 32], koff);

    float rm[8], rl[8];
    v8f o[4];
    #pragma unroll
    for (int r = 0; r < 8; ++r) { rm[r] = NEG_BIG; rl[r] = 0.f; }
    #pragma unroll
    for (int dt = 0; dt < 4; ++dt) o[dt] = (v8f){0.f,0.f,0.f,0.f,0.f,0.f,0.f,0.f};

    for (int kt = 0; kt <= qt; ++kt) {
        const int cur = kt & 1;
        if (kt < qt && w == 0) {         // async prefetch next K/V tile
            sa_tdm_load_2d(&Ks[1 - cur][0][0], kg + (size_t)(kt + 1) * 64 * SA_QKVN,
                           64, 64, SA_QKVN, 4, 3);
            sa_tdm_load_2d(&Vt[1 - cur][0][0], vg + (kt + 1) * 64,
                           64, 64, SA_T, 4, 3);
        }

        // ---- S = (Q K^T) * D^-1/2 ----
        v8f s[4];
        #pragma unroll
        for (int nt = 0; nt < 4; ++nt) {
            s[nt] = (v8f){0.f,0.f,0.f,0.f,0.f,0.f,0.f,0.f};
            #pragma unroll
            for (int c = 0; c < 2; ++c) {
                const v16h kb = *(const v16h*)&Ks[cur][nt * 16 + ln][c * 32 + koff2];
                s[nt] = __builtin_amdgcn_wmma_f32_16x16x32_f16(
                    false, qa[c], false, kb, (short)0, s[nt], false, false);
            }
            s[nt] = s[nt] * 0.125f;
        }

        if (kt == qt) {                  // causal mask on diagonal tile
            #pragma unroll
            for (int nt = 0; nt < 4; ++nt) {
                const int col = nt * 16 + ln;
                #pragma unroll
                for (int r = 0; r < 8; ++r)
                    if (col > w * 16 + r + hh * 8) s[nt][r] = NEG_BIG;
            }
        }

        // ---- online softmax (row spread over one 16-lane half) ----
        float alpha[8], rs[8];
        #pragma unroll
        for (int r = 0; r < 8; ++r) {
            float v = fmaxf(fmaxf(s[0][r], s[1][r]), fmaxf(s[2][r], s[3][r]));
            #pragma unroll
            for (int msk = 1; msk < 16; msk <<= 1)
                v = fmaxf(v, __shfl_xor(v, msk, 32));
            const float mnew = fmaxf(rm[r], v);
            alpha[r] = __expf(rm[r] - mnew);
            rm[r] = mnew;
            rs[r] = 0.f;
        }
        #pragma unroll
        for (int nt = 0; nt < 4; ++nt) {
            const int col = nt * 16 + ln;
            #pragma unroll
            for (int r = 0; r < 8; ++r) {
                const float p = __expf(s[nt][r] - rm[r]);
                rs[r] += p;
                Ps[w * 16 + r + hh * 8][col] = (_Float16)p;
            }
        }
        #pragma unroll
        for (int r = 0; r < 8; ++r) {
            #pragma unroll
            for (int msk = 1; msk < 16; msk <<= 1)
                rs[r] += __shfl_xor(rs[r], msk, 32);
            rl[r] = rl[r] * alpha[r] + rs[r];
        }
        #pragma unroll
        for (int dt = 0; dt < 4; ++dt)
            #pragma unroll
            for (int r = 0; r < 8; ++r)
                o[dt][r] = o[dt][r] * alpha[r];

        // ---- O += P V ----
        #pragma unroll
        for (int c = 0; c < 2; ++c) {
            const v16h pa = sa_load_afrag(&Ps[w * 16 + ln][c * 32], koff);
            #pragma unroll
            for (int dt = 0; dt < 4; ++dt) {
                const v16h vb = *(const v16h*)&Vt[cur][dt * 16 + ln][c * 32 + koff2];
                o[dt] = __builtin_amdgcn_wmma_f32_16x16x32_f16(
                    false, pa, false, vb, (short)0, o[dt], false, false);
            }
        }

        if (kt < qt) {
            if (w == 0) sa_wait_tdm();
            __syncthreads();
        }
    }

    #pragma unroll
    for (int dt = 0; dt < 4; ++dt) {
        const int d = dt * 16 + ln;
        #pragma unroll
        for (int r = 0; r < 8; ++r) {
            const int row = q0 + w * 16 + r + hh * 8;
            att[((size_t)b * SA_T + row) * SA_C + h * SA_D + d] =
                (_Float16)(o[dt][r] / rl[r]);
        }
    }
}

// ---------------------------------------------------------------------------
extern "C" void kernel_launch(void* const* d_in, const int* in_sizes, int n_in,
                              void* d_out, int out_size, void* d_ws, size_t ws_size,
                              hipStream_t stream) {
    (void)in_sizes; (void)n_in; (void)out_size; (void)ws_size;
    const float* x      = (const float*)d_in[0];
    const float* W_qkv  = (const float*)d_in[1];
    const float* b_qkv  = (const float*)d_in[2];
    const float* W_proj = (const float*)d_in[3];
    const float* b_proj = (const float*)d_in[4];
    float* out = (float*)d_out;

    const int M = SA_B * SA_T;   // 16384
    _Float16* p = (_Float16*)d_ws;
    _Float16* xh     = p;  p += (size_t)M * SA_C;
    _Float16* wqkvt  = p;  p += (size_t)SA_QKVN * SA_C;
    _Float16* wprojt = p;  p += (size_t)SA_C * SA_C;
    _Float16* qkvh   = p;  p += (size_t)M * SA_QKVN;
    _Float16* vtg    = p;  p += (size_t)SA_B * SA_H * SA_D * SA_T;
    _Float16* atth   = p;

    // one-time precision/layout prep
    sa_cvt<<<(M * SA_C / 4 + 255) / 256, 256, 0, stream>>>(x, xh, M * SA_C / 4);
    sa_wtrans<<<(SA_QKVN * SA_C + 255) / 256, 256, 0, stream>>>(W_qkv, wqkvt, SA_C, SA_QKVN);
    sa_wtrans<<<(SA_C * SA_C + 255) / 256, 256, 0, stream>>>(W_proj, wprojt, SA_C, SA_C);

    // qkv = x @ W_qkv + b   (f16 out)
    sa_gemm_wmma<_Float16><<<dim3(SA_QKVN / 64, M / 128), 128, 0, stream>>>(
        xh, wqkvt, b_qkv, qkvh, M, SA_QKVN, SA_C);

    // per-head V transpose, then causal flash attention
    sa_vtrans<<<dim3(SA_B * SA_H, SA_T / 64), 128, 0, stream>>>(qkvh, vtg);
    sa_attn_wmma<<<dim3(SA_B * SA_H, SA_T / 64), 128, 0, stream>>>(qkvh, vtg, atth);

    // out = att @ W_proj + b   (f32 out)
    sa_gemm_wmma<float><<<dim3(SA_C / 64, M / 128), 128, 0, stream>>>(
        atth, wprojt, b_proj, out, M, SA_C, SA_C);
}
